// LinearAttention_24507083391399
// MI455X (gfx1250) — compile-verified, matrix operand order checked by reading, not run on hardware
//
#include <hip/hip_runtime.h>

typedef __bf16 bf16;
typedef bf16 v16bf __attribute__((ext_vector_type(16)));
typedef bf16 v8bf  __attribute__((ext_vector_type(8)));
typedef float v8f  __attribute__((ext_vector_type(8)));
typedef int   v4i  __attribute__((ext_vector_type(4)));

#define T_SEQ   2048
#define D_MODEL 1024
#define N_HEADS 16
#define D_HEAD  64

// ---------------------------------------------------------------------------
// Optional CDNA5 async global->LDS copy (guarded; falls back to reg staging)
// Builtin signature (from clang diagnostic): (v4i addrspace(1)*, v4i addrspace(3)*, imm, imm)
// ---------------------------------------------------------------------------
#if defined(__has_builtin)
#if __has_builtin(__builtin_amdgcn_global_load_async_to_lds_b128) && \
    __has_builtin(__builtin_amdgcn_s_wait_asynccnt)
#define USE_ASYNC_LDS 1
#endif
#endif

#if defined(USE_ASYNC_LDS)
typedef __attribute__((address_space(1))) v4i* as1_v4i;
typedef __attribute__((address_space(3))) v4i* as3_v4i;
#define ASYNC_CP16(gsrc, ldst)                                              \
  __builtin_amdgcn_global_load_async_to_lds_b128(                           \
      (as1_v4i)(unsigned long long)(uintptr_t)(gsrc),                       \
      (as3_v4i)(unsigned int)(uintptr_t)(ldst), 0, 0)
#define WAIT_CP() __builtin_amdgcn_s_wait_asynccnt(0)
#else
#define WAIT_CP()
#endif

// ---------------------------------------------------------------------------
// WMMA helpers
// ---------------------------------------------------------------------------
__device__ __forceinline__ v8f wmma_bf16(v16bf a, v16bf b, v8f c) {
  return __builtin_amdgcn_wmma_f32_16x16x32_bf16(false, a, false, b,
                                                 (short)0, c, false, false);
}

// Per-lane fragment: two contiguous 8-element K runs at p[0..7] and p[16..23]
// (caller passes p already offset by khalf = (lane<16 ? 0 : 8)).
__device__ __forceinline__ v16bf ldfrag(const bf16* p) {
  v8bf lo = *(const v8bf*)p;
  v8bf hi = *(const v8bf*)(p + 16);
  v16bf r;
#pragma unroll
  for (int i = 0; i < 8; ++i) { r[i] = lo[i]; r[i + 8] = hi[i]; }
  return r;
}

// ---------------------------------------------------------------------------
// fp32 -> bf16 cast (vector of 4 per thread)
// ---------------------------------------------------------------------------
__global__ __launch_bounds__(256) void cvt4(bf16* __restrict__ dst,
                                            const float* __restrict__ src,
                                            int n) {
  int i = (blockIdx.x * 256 + threadIdx.x) * 4;
  if (i + 3 < n) {
    float4 f = *(const float4*)(src + i);
    dst[i + 0] = (bf16)f.x;
    dst[i + 1] = (bf16)f.y;
    dst[i + 2] = (bf16)f.z;
    dst[i + 3] = (bf16)f.w;
  }
}

// ---------------------------------------------------------------------------
// C(f32, MxN) = A(bf16, MxK rm) * B(bf16, NxK rm)^T
// 128x128 block tile, 4 waves (blockDim 128), each wave 64x64 (4x4 WMMA),
// BK=32, double-buffered LDS, one barrier per K-step.
// blockIdx.z selects among up to 3 (B, C) pairs sharing the same A.
// ---------------------------------------------------------------------------
#define BM 128
#define BN 128
#define BK 32
#define LDP (BK + 8)   // 80-byte padded rows, keeps 16B alignment

struct Stage { v8bf a[4], b[4]; };

// Issue the copy of one 128x32 A tile + 128x32 B tile for K offset k0.
__device__ __forceinline__ void issue_copy(Stage& st,
                                           bf16 (*As)[LDP], bf16 (*Bs)[LDP],
                                           const bf16* A, const bf16* B,
                                           int K, int k0, int blockM, int blockN,
                                           int tid) {
#pragma unroll
  for (int i = 0; i < 4; ++i) {
    int c = i * 128 + tid;
    int r = c >> 2;
    int col = (c & 3) << 3;
#if defined(USE_ASYNC_LDS)
    ASYNC_CP16(&A[(size_t)(blockM + r) * K + k0 + col], &As[r][col]);
    ASYNC_CP16(&B[(size_t)(blockN + r) * K + k0 + col], &Bs[r][col]);
#else
    st.a[i] = *(const v8bf*)&A[(size_t)(blockM + r) * K + k0 + col];
    st.b[i] = *(const v8bf*)&B[(size_t)(blockN + r) * K + k0 + col];
#endif
  }
}

__device__ __forceinline__ void commit_copy(Stage& st,
                                            bf16 (*As)[LDP], bf16 (*Bs)[LDP],
                                            int tid) {
#if !defined(USE_ASYNC_LDS)
#pragma unroll
  for (int i = 0; i < 4; ++i) {
    int c = i * 128 + tid;
    int r = c >> 2;
    int col = (c & 3) << 3;
    *(v8bf*)&As[r][col] = st.a[i];
    *(v8bf*)&Bs[r][col] = st.b[i];
  }
#endif
}

__global__ __launch_bounds__(128) void gemm_nt_multi(
    float* __restrict__ C0, float* __restrict__ C1, float* __restrict__ C2,
    const bf16* __restrict__ A,
    const bf16* __restrict__ B0, const bf16* __restrict__ B1,
    const bf16* __restrict__ B2,
    int M, int N, int K) {
  __shared__ bf16 As[2][BM][LDP];
  __shared__ bf16 Bs[2][BN][LDP];

  const bf16* Bm[3] = {B0, B1, B2};
  float*      Cm[3] = {C0, C1, C2};
  const bf16* B = Bm[blockIdx.z];
  float*      C = Cm[blockIdx.z];

  const int tid  = threadIdx.x;
  const int wave = tid >> 5;
  const int lane = tid & 31;
  const int waveM = wave & 1;     // 2 M-slices of 64
  const int waveN = wave >> 1;    // 2 N-slices of 64
  const int blockM = blockIdx.x * BM;
  const int blockN = blockIdx.y * BN;

  const int lrow  = lane & 15;
  const int khalf = (lane < 16) ? 0 : 8;
  const int mhi   = (lane >> 4) << 3;

  v8f acc[4][4] = {};
  Stage st;

  // prologue: tile k0 = 0 into buffer 0
  issue_copy(st, As[0], Bs[0], A, B, K, 0, blockM, blockN, tid);
  commit_copy(st, As[0], Bs[0], tid);
  WAIT_CP();
  __syncthreads();

  for (int k0 = 0; k0 < K; k0 += BK) {
    const int buf = (k0 / BK) & 1;
    const bool more = (k0 + BK) < K;

    if (more)
      issue_copy(st, As[buf ^ 1], Bs[buf ^ 1], A, B, K, k0 + BK,
                 blockM, blockN, tid);

    v16bf af[4];
#pragma unroll
    for (int mt = 0; mt < 4; ++mt)
      af[mt] = ldfrag(&As[buf][waveM * 64 + mt * 16 + lrow][khalf]);
#pragma unroll
    for (int nt = 0; nt < 4; ++nt) {
      v16bf bb = ldfrag(&Bs[buf][waveN * 64 + nt * 16 + lrow][khalf]);
#pragma unroll
      for (int mt = 0; mt < 4; ++mt)
        acc[mt][nt] = wmma_bf16(af[mt], bb, acc[mt][nt]);
    }

    if (more) commit_copy(st, As[buf ^ 1], Bs[buf ^ 1], tid);
    WAIT_CP();
    __syncthreads();
  }

#pragma unroll
  for (int mt = 0; mt < 4; ++mt) {
#pragma unroll
    for (int nt = 0; nt < 4; ++nt) {
      int gm = blockM + waveM * 64 + mt * 16 + mhi;
      int gn = blockN + waveN * 64 + nt * 16 + lrow;
#pragma unroll
      for (int r = 0; r < 8; ++r)
        C[(size_t)(gm + r) * N + gn] = acc[mt][nt][r];
    }
  }
}

// ---------------------------------------------------------------------------
// Per-(token,head) LayerNorm over 64 elements; one wave per row; bf16 out
// ---------------------------------------------------------------------------
__global__ __launch_bounds__(256) void ln64(bf16* __restrict__ dst,
                                            const float* __restrict__ src,
                                            const float* __restrict__ gamma,
                                            const float* __restrict__ beta,
                                            int nrows) {
  int row  = blockIdx.x * 8 + (threadIdx.x >> 5);
  int lane = threadIdx.x & 31;
  if (row >= nrows) return;
  const float* p = src + (size_t)row * 64;
  float a = p[lane], b = p[lane + 32];
  float s = a + b;
#pragma unroll
  for (int off = 16; off; off >>= 1) s += __shfl_xor(s, off, 32);
  float mu = s * (1.0f / 64.0f);
  float da = a - mu, db = b - mu;
  float v = da * da + db * db;
#pragma unroll
  for (int off = 16; off; off >>= 1) v += __shfl_xor(v, off, 32);
  float rinv = rsqrtf(v * (1.0f / 64.0f) + 1e-5f);
  bf16* q = dst + (size_t)row * 64;
  q[lane]      = (bf16)(da * rinv * gamma[lane] + beta[lane]);
  q[lane + 32] = (bf16)(db * rinv * gamma[lane + 32] + beta[lane + 32]);
}

// ---------------------------------------------------------------------------
// Chunked causal linear attention, one workgroup per head, chunk = 64.
// Per chunk (all via WMMA):  P = tril(q k^T) ; o = P v + q S ; S += k^T v
// ---------------------------------------------------------------------------
#define LDA 72   // 144-byte rows -> 16B aligned fragment loads

__global__ __launch_bounds__(256) void linattn_kernel(bf16* __restrict__ O,
                                                      const bf16* __restrict__ Q,
                                                      const bf16* __restrict__ Kn,
                                                      const bf16* __restrict__ V,
                                                      int T) {
  __shared__ bf16  qs [64][LDA];   // q chunk, row-major (t, d)
  __shared__ bf16  ks [64][LDA];   // k chunk, row-major (t, d)
  __shared__ bf16  kT [64][LDA];   // k transposed (d, t)
  __shared__ bf16  vT [64][LDA];   // v transposed (e, t)
  __shared__ bf16  Ps [64][LDA];   // masked scores (t, t')
  __shared__ bf16  SbT[64][LDA];   // bf16 snapshot of S, transposed (e, d)
  __shared__ float Sf [64][64];    // running state S[d][e], f32

  const int h    = blockIdx.x;
  const int tid  = threadIdx.x;
  const int wave = tid >> 5;
  const int lane = tid & 31;
  const int tm   = wave & 3;       // M-tile (rows of chunk)
  const int tn0  = wave >> 2;      // owns N-tiles tn0 and tn0+2
  const int lrow  = lane & 15;
  const int khalf = (lane < 16) ? 0 : 8;
  const int mhi   = (lane >> 4) << 3;

  for (int i = tid; i < 64 * 64; i += 256) ((float*)Sf)[i] = 0.0f;
  __syncthreads();

  for (int c = 0; c < T; c += 64) {
    // ---- stage 0: load q/k/v chunk, build transposes, snapshot S ----
    {
      int r  = tid >> 2;            // 0..63 (row of chunk)
      int cc = (tid & 3) << 4;      // 0,16,32,48
      size_t base = ((size_t)(c + r) * N_HEADS + h) * D_HEAD;
      v8bf q0 = *(const v8bf*)&Q[base + cc];
      v8bf q1 = *(const v8bf*)&Q[base + cc + 8];
      *(v8bf*)&qs[r][cc]     = q0;
      *(v8bf*)&qs[r][cc + 8] = q1;
      v8bf k0 = *(const v8bf*)&Kn[base + cc];
      v8bf k1 = *(const v8bf*)&Kn[base + cc + 8];
      *(v8bf*)&ks[r][cc]     = k0;
      *(v8bf*)&ks[r][cc + 8] = k1;
      v8bf v0 = *(const v8bf*)&V[base + cc];
      v8bf v1 = *(const v8bf*)&V[base + cc + 8];
#pragma unroll
      for (int j = 0; j < 8; ++j) {
        kT[cc + j][r]     = k0[j];
        kT[cc + 8 + j][r] = k1[j];
        vT[cc + j][r]     = v0[j];
        vT[cc + 8 + j][r] = v1[j];
      }
    }
    for (int i = tid; i < 64 * 64; i += 256) {
      int d = i >> 6, e = i & 63;
      SbT[e][d] = (bf16)Sf[d][e];
    }
    __syncthreads();

    // ---- stage 1: P = tril(q k^T) ----
#pragma unroll
    for (int t = 0; t < 2; ++t) {
      int tn = tn0 + t * 2;
      v8f p = {};
#pragma unroll
      for (int kk = 0; kk < 2; ++kk) {
        v16bf a = ldfrag(&qs[tm * 16 + lrow][kk * 32 + khalf]);
        v16bf b = ldfrag(&ks[tn * 16 + lrow][kk * 32 + khalf]);
        p = wmma_bf16(a, b, p);
      }
      int n = tn * 16 + lrow;
#pragma unroll
      for (int r = 0; r < 8; ++r) {
        int m = tm * 16 + mhi + r;
        float val = (n <= m) ? p[r] : 0.0f;   // causal (inclusive)
        Ps[m][n] = (bf16)val;
      }
    }
    __syncthreads();

    // ---- stage 2: o = P v + q S ----
#pragma unroll
    for (int t = 0; t < 2; ++t) {
      int te = tn0 + t * 2;
      v8f o = {};
#pragma unroll
      for (int kk = 0; kk < 2; ++kk) {
        v16bf a  = ldfrag(&Ps[tm * 16 + lrow][kk * 32 + khalf]);
        v16bf b  = ldfrag(&vT[te * 16 + lrow][kk * 32 + khalf]);
        o = wmma_bf16(a, b, o);
        v16bf a2 = ldfrag(&qs[tm * 16 + lrow][kk * 32 + khalf]);
        v16bf b2 = ldfrag(&SbT[te * 16 + lrow][kk * 32 + khalf]);
        o = wmma_bf16(a2, b2, o);
      }
#pragma unroll
      for (int r = 0; r < 8; ++r) {
        int m = tm * 16 + mhi + r;
        O[((size_t)(c + m) * N_HEADS + h) * D_HEAD + te * 16 + lrow] = (bf16)o[r];
      }
    }

    // ---- stage 3: S += k^T v (each wave owns disjoint S tiles) ----
#pragma unroll
    for (int t = 0; t < 2; ++t) {
      int te = tn0 + t * 2;
      v8f s = {};
#pragma unroll
      for (int r = 0; r < 8; ++r) s[r] = Sf[tm * 16 + mhi + r][te * 16 + lrow];
#pragma unroll
      for (int kk = 0; kk < 2; ++kk) {
        v16bf a = ldfrag(&kT[tm * 16 + lrow][kk * 32 + khalf]);
        v16bf b = ldfrag(&vT[te * 16 + lrow][kk * 32 + khalf]);
        s = wmma_bf16(a, b, s);
      }
#pragma unroll
      for (int r = 0; r < 8; ++r) Sf[tm * 16 + mhi + r][te * 16 + lrow] = s[r];
    }
    __syncthreads();
  }
}

// ---------------------------------------------------------------------------
// Host-side orchestration
// ---------------------------------------------------------------------------
extern "C" void kernel_launch(void* const* d_in, const int* in_sizes, int n_in,
                              void* d_out, int out_size, void* d_ws, size_t ws_size,
                              hipStream_t stream) {
  const float* x  = (const float*)d_in[0];
  const float* Wq = (const float*)d_in[1];
  const float* Wk = (const float*)d_in[2];
  const float* Wv = (const float*)d_in[3];
  const float* Wo = (const float*)d_in[4];
  const float* qg = (const float*)d_in[5];
  const float* qb = (const float*)d_in[6];
  const float* kg = (const float*)d_in[7];
  const float* kb = (const float*)d_in[8];

  const int T = T_SEQ, DM = D_MODEL;
  const size_t NX = (size_t)T * DM;      // 2M elements
  const size_t NW = (size_t)DM * DM;     // 1M elements

  char* ws = (char*)d_ws;
  size_t off = 0;
  auto take = [&](size_t bytes) -> char* {
    char* p = ws + off;
    off += (bytes + 255) & ~(size_t)255;
    return p;
  };
  bf16* xb  = (bf16*)take(NX * sizeof(bf16));
  bf16* wqb = (bf16*)take(NW * sizeof(bf16));
  bf16* wkb = (bf16*)take(NW * sizeof(bf16));
  bf16* wvb = (bf16*)take(NW * sizeof(bf16));
  bf16* wob = (bf16*)take(NW * sizeof(bf16));
  float* qf = (float*)take(NX * sizeof(float));
  float* kf = (float*)take(NX * sizeof(float));
  float* vf = (float*)take(NX * sizeof(float));
  bf16* qn  = (bf16*)take(NX * sizeof(bf16));
  bf16* kn  = (bf16*)take(NX * sizeof(bf16));
  bf16* vb  = (bf16*)take(NX * sizeof(bf16));
  bf16* ob  = (bf16*)take(NX * sizeof(bf16));

  // 1) casts to bf16
  cvt4<<<dim3((NX / 4 + 255) / 256), 256, 0, stream>>>(xb, x, (int)NX);
  cvt4<<<dim3((NW / 4 + 255) / 256), 256, 0, stream>>>(wqb, Wq, (int)NW);
  cvt4<<<dim3((NW / 4 + 255) / 256), 256, 0, stream>>>(wkb, Wk, (int)NW);
  cvt4<<<dim3((NW / 4 + 255) / 256), 256, 0, stream>>>(wvb, Wv, (int)NW);
  cvt4<<<dim3((NW / 4 + 255) / 256), 256, 0, stream>>>(wob, Wo, (int)NW);

  // 2) fused QKV projections (C = A * B^T), z selects q/k/v
  dim3 gq(T / BM, DM / BN, 3);
  gemm_nt_multi<<<gq, 128, 0, stream>>>(qf, kf, vf, xb, wqb, wkb, wvb,
                                        T, DM, DM);

  // 3) per-head LayerNorm (q, k) + bf16 cast of v
  int nrows = T * N_HEADS;
  ln64<<<dim3(nrows / 8), 256, 0, stream>>>(qn, qf, qg, qb, nrows);
  ln64<<<dim3(nrows / 8), 256, 0, stream>>>(kn, kf, kg, kb, nrows);
  cvt4<<<dim3((NX / 4 + 255) / 256), 256, 0, stream>>>(vb, vf, (int)NX);

  // 4) chunked causal linear attention (one WG per head)
  linattn_kernel<<<dim3(N_HEADS), 256, 0, stream>>>(ob, qn, kn, vb, T);

  // 5) output projection into d_out
  dim3 go(T / BM, DM / BN, 1);
  gemm_nt_multi<<<go, 128, 0, stream>>>((float*)d_out, nullptr, nullptr,
                                        ob, wob, wob, wob, T, DM, DM);
}
